// StandardizedMamba_2164663517861
// MI455X (gfx1250) — matrix operand order repaired and checked by reading, not compiled, not run on hardware
//
#include <hip/hip_runtime.h>
#include <hip/hip_bf16.h>
#include <math.h>

// ---------------- model dims ----------------
#define D_MODEL 768
#define D_INNER 1536       // 2*D_MODEL
#define XP_COLS 3072       // 2*D_INNER
#define D_STATE 16
#define BATCH   2
#define SEQ     2048
#define M_TOK   (BATCH*SEQ)   // 4096
#define VOCAB   32000
#define LN_EPS  1e-5f

// chunk-parallel scan decomposition
#define NCHUNK  16
#define CLEN    (SEQ / NCHUNK)   // 128

typedef __attribute__((ext_vector_type(16))) _Float16 v16h;
typedef __attribute__((ext_vector_type(8)))  float    v8f;

union FragU { v16h v; unsigned int u[8]; };

// =====================================================================
// Tiled NT GEMM:  C[M,N] = A[M,K] * B[N,K]^T   (both K-contiguous, f32)
// f32 -> f16 conversion on LDS stage-in, f32 accumulation via
// v_wmma_f32_16x16x32_f16.
//   block tile 128x128, wave tile 32x64 (8 WMMA / wave / K-step),
//   double-buffered LDS + register-staged global prefetch.
// mode 0: plain store.  mode 1: C = clip(softplus(C + b1[n] + b2[n]),0,1)
// Requires: M % 128 == 0, K % 32 == 0.  N arbitrary (bounds-checked).
// =====================================================================
#define BM 128
#define BN 128
#define BK 32
#define LDT 34  // LDS row stride in halves (17 dwords -> conflict-free)

__device__ __forceinline__ v8f wmma_f16(const FragU& a, const FragU& b, v8f c) {
    return __builtin_amdgcn_wmma_f32_16x16x32_f16(
        false, a.v, false, b.v, (short)0, c, false, false);
}

__global__ __launch_bounds__(256)
void gemm_nt_wmma(const float* __restrict__ A, const float* __restrict__ B,
                  float* __restrict__ C, int M, int N, int K,
                  int mode, const float* __restrict__ bias1,
                  const float* __restrict__ bias2)
{
    __shared__ _Float16 As[2][BM * LDT];
    __shared__ _Float16 Bs[2][BN * LDT];

    const int tid   = threadIdx.x;
    const int wid   = tid >> 5;
    const int lane  = tid & 31;
    const int hi    = lane >> 4;    // half-wave select
    const int ln16  = lane & 15;
    const int waveM = wid & 3;      // 4 wave-rows * 32  -> block M = 128
    const int waveN = wid >> 2;     // 2 wave-cols * 64  -> block N = 128
    const int tileM = blockIdx.y * BM;
    const int tileN = blockIdx.x * BN;

    // staging: each thread moves 2 rows x 8 floats for A and for B
    const int sr = tid >> 2;          // 0..63  (rows sr and sr+64)
    const int sc = (tid & 3) * 8;     // 0,8,16,24

    bool bValid[2];
    #pragma unroll
    for (int rr = 0; rr < 2; ++rr) bValid[rr] = (tileN + sr + rr * 64) < N;

    v8f acc[2][4] = {};               // eight 16x16 f32 accumulators

    float4 aReg[2][2], bReg[2][2];

    // ---- issue global loads for K-step kt into registers ----
    auto ldreg = [&](int kt) {
        const int k0 = kt * BK;
        #pragma unroll
        for (int rr = 0; rr < 2; ++rr) {
            const int r = sr + rr * 64;
            const float* srcA = A + (size_t)(tileM + r) * K + k0 + sc;
            aReg[rr][0] = ((const float4*)srcA)[0];
            aReg[rr][1] = ((const float4*)srcA)[1];
            if (bValid[rr]) {
                const float* srcB = B + (size_t)(tileN + r) * K + k0 + sc;
                bReg[rr][0] = ((const float4*)srcB)[0];
                bReg[rr][1] = ((const float4*)srcB)[1];
            }
        }
    };

    // ---- convert staged registers to f16 and store into LDS buffer ----
    auto cvtstore = [&](int buf) {
        #pragma unroll
        for (int rr = 0; rr < 2; ++rr) {
            const int r = sr + rr * 64;
            _Float16* da = &As[buf][r * LDT + sc];
            da[0] = (_Float16)aReg[rr][0].x; da[1] = (_Float16)aReg[rr][0].y;
            da[2] = (_Float16)aReg[rr][0].z; da[3] = (_Float16)aReg[rr][0].w;
            da[4] = (_Float16)aReg[rr][1].x; da[5] = (_Float16)aReg[rr][1].y;
            da[6] = (_Float16)aReg[rr][1].z; da[7] = (_Float16)aReg[rr][1].w;
            _Float16* db = &Bs[buf][r * LDT + sc];
            if (bValid[rr]) {
                db[0] = (_Float16)bReg[rr][0].x; db[1] = (_Float16)bReg[rr][0].y;
                db[2] = (_Float16)bReg[rr][0].z; db[3] = (_Float16)bReg[rr][0].w;
                db[4] = (_Float16)bReg[rr][1].x; db[5] = (_Float16)bReg[rr][1].y;
                db[6] = (_Float16)bReg[rr][1].z; db[7] = (_Float16)bReg[rr][1].w;
            } else {
                #pragma unroll
                for (int q = 0; q < 8; ++q) db[q] = (_Float16)0.0f;
            }
        }
    };

    // ---- 8 WMMAs on buffer buf: 2 A-frags x 4 B-frags, all frags loaded
    //      up-front into distinct registers so WMMAs can issue back-to-back
    auto compute = [&](int buf) {
        FragU afr[2];
        #pragma unroll
        for (int i = 0; i < 2; ++i) {
            const _Float16* arow = &As[buf][(waveM * 32 + i * 16 + ln16) * LDT];
            #pragma unroll
            for (int p = 0; p < 8; ++p) {
                const int kk = 2 * p + 16 * hi;  // contiguous k order (k = 16*hi + s), matches B
                afr[i].u[p] = *(const unsigned int*)&arow[kk];
            }
        }
        FragU bfr[4];
        #pragma unroll
        for (int j = 0; j < 4; ++j) {
            const _Float16* brow =
                &Bs[buf][(waveN * 64 + j * 16 + ln16) * LDT + 16 * hi];
            #pragma unroll
            for (int q = 0; q < 8; ++q)
                bfr[j].u[q] = *(const unsigned int*)&brow[2 * q];
        }
        #pragma unroll
        for (int j = 0; j < 4; ++j) {
            acc[0][j] = wmma_f16(afr[0], bfr[j], acc[0][j]);
            acc[1][j] = wmma_f16(afr[1], bfr[j], acc[1][j]);
        }
    };

    // ---- software-pipelined main loop ----
    const int KT = K / BK;
    ldreg(0);
    cvtstore(0);
    __syncthreads();
    for (int kt = 0; kt < KT; ++kt) {
        const int  cur  = kt & 1;
        const bool more = (kt + 1) < KT;
        if (more) ldreg(kt + 1);      // global loads in flight during WMMAs
        compute(cur);
        if (more) {
            cvtstore(cur ^ 1);        // fill the other buffer
            __syncthreads();          // single barrier per K-step
        }
    }

    // ---- epilogue: D layout = { M = r + 8*hi, N = ln16 } per vgpr r ----
    #pragma unroll
    for (int j = 0; j < 4; ++j) {
        const int n = tileN + waveN * 64 + j * 16 + ln16;
        if (n < N) {
            float bsum = 0.0f;
            if (mode == 1) bsum = bias1[n] + bias2[n];
            #pragma unroll
            for (int i = 0; i < 2; ++i) {
                #pragma unroll
                for (int r = 0; r < 8; ++r) {
                    const int m = tileM + waveM * 32 + i * 16 + r + 8 * hi;
                    float v = acc[i][j][r];
                    if (mode == 1) {
                        float t  = v + bsum;
                        float sp = (t > 15.0f) ? t : log1pf(expf(t)); // softplus
                        v = fminf(fmaxf(sp, 0.0f), 1.0f);             // clip
                    }
                    C[(size_t)m * N + n] = v;
                }
            }
        }
    }
}

// =====================================================================
// Embedding gather: x[tok, c] = emb[ids[tok], c]
// =====================================================================
__global__ void embed_kernel(const int* __restrict__ ids,
                             const float* __restrict__ emb,
                             float* __restrict__ x, int ntok)
{
    int i = blockIdx.x * blockDim.x + threadIdx.x;
    if (i < ntok * D_MODEL) {
        int t = i / D_MODEL;
        int c = i - t * D_MODEL;
        x[i] = emb[(size_t)ids[t] * D_MODEL + c];
    }
}

// =====================================================================
// Chunk-parallel selective scan.
// Recurrence per (b,d,s):  h_t = f_t * h_{t-1} + g_t   (linear in h)
//   f_t = 1 + A[d,s]*dt_t*0.1,  g_t = B_t[s]*u_t*0.1
// Pass 1: per chunk, scan from h=0 -> chunk product F[s] and endpoint h0[s]
// Pass 2: per channel, combine the NCHUNK summaries sequentially -> hstart
// Pass 3: rescan each chunk from hstart, emit yg = (h.C + D*u)*sigmoid(gate)
// Summary layout: [(b*NCHUNK + c)*D_INNER + d]*16 + s in pass 1/3; pass 2 uses c*BATCH+b   (coalesced in d)
// =====================================================================
__global__ __launch_bounds__(256)
void scan_pass1(const float* __restrict__ xp,   // [M, 3072]
                const float* __restrict__ dtc,  // [M, 1536]
                const float* __restrict__ Bm,   // [M, 16]
                const float* __restrict__ Aarr, // [1536, 16]
                float* __restrict__ Fbuf,       // [B, NC, 1536, 16]
                float* __restrict__ H0buf)      // [B, NC, 1536, 16]
{
    const int bpb  = D_INNER / 256;                 // 6 blocks per (b,c)
    const int cidx = blockIdx.x / bpb;              // 0 .. BATCH*NCHUNK-1
    const int d    = (blockIdx.x % bpb) * 256 + threadIdx.x;
    const int b    = cidx / NCHUNK;
    const int c    = cidx % NCHUNK;

    float a01[D_STATE], h[D_STATE], F[D_STATE];
    #pragma unroll
    for (int s = 0; s < D_STATE; ++s) {
        a01[s] = Aarr[(size_t)d * D_STATE + s] * 0.1f;
        h[s] = 0.0f; F[s] = 1.0f;
    }

    const int t0 = c * CLEN;
    for (int t = t0; t < t0 + CLEN; ++t) {
        const size_t row = (size_t)b * SEQ + t;
        const float u  = xp[row * XP_COLS + d];
        const float dt = dtc[row * D_INNER + d];
        const float* Bt = Bm + row * D_STATE;       // block-uniform address
        const float u01 = u * 0.1f;
        #pragma unroll
        for (int s = 0; s < D_STATE; ++s) {
            const float f = fmaf(a01[s], dt, 1.0f);
            F[s] *= f;
            h[s] = fmaf(h[s], f, Bt[s] * u01);
        }
    }
    float* fd = Fbuf  + ((size_t)cidx * D_INNER + d) * D_STATE;
    float* hd = H0buf + ((size_t)cidx * D_INNER + d) * D_STATE;
    #pragma unroll
    for (int s = 0; s < D_STATE; ++s) { fd[s] = F[s]; hd[s] = h[s]; }
}

__global__ __launch_bounds__(256)
void scan_pass2(const float* __restrict__ Fbuf,   // [NC, B, 1536, 16]
                const float* __restrict__ H0buf,  // [NC, B, 1536, 16]
                float* __restrict__ HSbuf)        // [NC, B, 1536, 16]
{
    const int bpb = D_INNER / 256;
    const int b   = blockIdx.x / bpb;
    const int d   = (blockIdx.x % bpb) * 256 + threadIdx.x;

    float h[D_STATE];
    #pragma unroll
    for (int s = 0; s < D_STATE; ++s) h[s] = 0.0f;

    for (int c = 0; c < NCHUNK; ++c) {
        const size_t off = ((size_t)(c * BATCH + b) * D_INNER + d) * D_STATE;
        float* hs = HSbuf + off;
        #pragma unroll
        for (int s = 0; s < D_STATE; ++s) hs[s] = h[s];   // chunk-start state
        const float* fd = Fbuf + off;
        const float* hd = H0buf + off;
        #pragma unroll
        for (int s = 0; s < D_STATE; ++s)
            h[s] = fmaf(fd[s], h[s], hd[s]);
    }
}

__global__ __launch_bounds__(256)
void scan_pass3(const float* __restrict__ xp,    // [M, 3072] (ssm | gate)
                const float* __restrict__ dtc,   // [M, 1536]
                const float* __restrict__ Bm,    // [M, 16]
                const float* __restrict__ Cm,    // [M, 16]
                const float* __restrict__ Aarr,  // [1536, 16]
                const float* __restrict__ Darr,  // [1536]
                const float* __restrict__ HSbuf, // [B, NC, 1536, 16]
                float* __restrict__ yg)          // [M, 1536]
{
    const int bpb  = D_INNER / 256;
    const int cidx = blockIdx.x / bpb;
    const int d    = (blockIdx.x % bpb) * 256 + threadIdx.x;
    const int b    = cidx / NCHUNK;
    const int c    = cidx % NCHUNK;

    float a01[D_STATE], h[D_STATE];
    const float* hs = HSbuf + ((size_t)cidx * D_INNER + d) * D_STATE;
    #pragma unroll
    for (int s = 0; s < D_STATE; ++s) {
        a01[s] = Aarr[(size_t)d * D_STATE + s] * 0.1f;
        h[s]   = hs[s];
    }
    const float Dd = Darr[d];

    const int t0 = c * CLEN;
    for (int t = t0; t < t0 + CLEN; ++t) {
        const size_t row = (size_t)b * SEQ + t;
        const float u  = xp[row * XP_COLS + d];
        const float g  = xp[row * XP_COLS + D_INNER + d];
        const float dt = dtc[row * D_INNER + d];
        const float* Bt = Bm + row * D_STATE;       // block-uniform
        const float* Ct = Cm + row * D_STATE;
        const float u01 = u * 0.1f;
        float y = Dd * u;
        #pragma unroll
        for (int s = 0; s < D_STATE; ++s) {
            const float f = fmaf(a01[s], dt, 1.0f);
            h[s] = fmaf(h[s], f, Bt[s] * u01);
            y    = fmaf(h[s], Ct[s], y);
        }
        const float gate = 1.0f / (1.0f + expf(-g));
        yg[row * D_INNER + d] = y * gate;
    }
}

// =====================================================================
// LayerNorm over D_MODEL=768; one block per row; optional residual add.
// Safe for dst == src / dst == res (row fully read before write).
// =====================================================================
__global__ __launch_bounds__(256)
void ln_kernel(const float* __restrict__ src, const float* __restrict__ res,
               const float* __restrict__ g, const float* __restrict__ b,
               float* __restrict__ dst)
{
    __shared__ float red[256];
    const int row = blockIdx.x;
    const int tid = threadIdx.x;
    float v[3];
    float s = 0.0f;
    #pragma unroll
    for (int j = 0; j < 3; ++j) {
        const int c = tid + j * 256;
        float t = src[(size_t)row * D_MODEL + c];
        if (res) t += res[(size_t)row * D_MODEL + c];
        v[j] = t; s += t;
    }
    red[tid] = s; __syncthreads();
    for (int off = 128; off > 0; off >>= 1) {
        if (tid < off) red[tid] += red[tid + off];
        __syncthreads();
    }
    const float mean = red[0] * (1.0f / D_MODEL);
    __syncthreads();
    float s2 = 0.0f;
    #pragma unroll
    for (int j = 0; j < 3; ++j) { const float dv = v[j] - mean; s2 += dv * dv; }
    red[tid] = s2; __syncthreads();
    for (int off = 128; off > 0; off >>= 1) {
        if (tid < off) red[tid] += red[tid + off];
        __syncthreads();
    }
    const float rstd = rsqrtf(red[0] * (1.0f / D_MODEL) + LN_EPS);
    #pragma unroll
    for (int j = 0; j < 3; ++j) {
        const int c = tid + j * 256;
        dst[(size_t)row * D_MODEL + c] = (v[j] - mean) * rstd * g[c] + b[c];
    }
}

// =====================================================================
extern "C" void kernel_launch(void* const* d_in, const int* in_sizes, int n_in,
                              void* d_out, int out_size, void* d_ws, size_t ws_size,
                              hipStream_t stream)
{
    const int*   ids     = (const int*)d_in[0];
    const float* emb     = (const float*)d_in[1];
    const float* W_in    = (const float*)d_in[2];   // [2, 3072, 768]
    const float* W_dt    = (const float*)d_in[3];   // [2, 1536, 768]
    const float* b_dt    = (const float*)d_in[4];   // [2, 1536]
    const float* dt_bias = (const float*)d_in[5];   // [2, 1536]
    const float* W_B     = (const float*)d_in[6];   // [2, 16, 768]
    const float* W_C     = (const float*)d_in[7];   // [2, 16, 768]
    const float* Aarr    = (const float*)d_in[8];   // [2, 1536, 16]
    const float* Darr    = (const float*)d_in[9];   // [2, 1536]
    const float* W_out   = (const float*)d_in[10];  // [2, 768, 1536]
    const float* ln_g    = (const float*)d_in[11];  // [2, 768]
    const float* ln_b    = (const float*)d_in[12];
    const float* fin_g   = (const float*)d_in[13];  // [768]
    const float* fin_b   = (const float*)d_in[14];
    float* out = (float*)d_out;                     // [4096, 32000] f32

    // workspace carve-up (f32)
    const size_t SUMSZ = (size_t)NCHUNK * BATCH * D_INNER * D_STATE; // 786432
    float* ws   = (float*)d_ws;
    float* x    = ws;                               // 4096*768
    float* xp   = x    + (size_t)M_TOK * D_MODEL;   // 4096*3072
    float* dtc  = xp   + (size_t)M_TOK * XP_COLS;   // 4096*1536
    float* Bm   = dtc  + (size_t)M_TOK * D_INNER;   // 4096*16
    float* Cm   = Bm   + (size_t)M_TOK * D_STATE;   // 4096*16
    float* yg   = Cm   + (size_t)M_TOK * D_STATE;   // 4096*1536
    float* proj = yg   + (size_t)M_TOK * D_INNER;   // 4096*768
    float* Fbuf = proj + (size_t)M_TOK * D_MODEL;   // scan summaries
    float* H0b  = Fbuf + SUMSZ;
    float* HSb  = H0b  + SUMSZ;

    const int mtiles = M_TOK / BM;   // 32

    embed_kernel<<<(M_TOK * D_MODEL + 255) / 256, 256, 0, stream>>>(ids, emb, x, M_TOK);

    const int scanBlocks = BATCH * NCHUNK * (D_INNER / 256);   // 192
    const int combBlocks = BATCH * (D_INNER / 256);            // 12

    for (int l = 0; l < 2; ++l) {
        const float* Wi  = W_in  + (size_t)l * XP_COLS * D_MODEL;
        const float* Wd  = W_dt  + (size_t)l * D_INNER * D_MODEL;
        const float* bd  = b_dt  + (size_t)l * D_INNER;
        const float* db  = dt_bias + (size_t)l * D_INNER;
        const float* WBl = W_B   + (size_t)l * D_STATE * D_MODEL;
        const float* WCl = W_C   + (size_t)l * D_STATE * D_MODEL;
        const float* Al  = Aarr  + (size_t)l * D_INNER * D_STATE;
        const float* Dl  = Darr  + (size_t)l * D_INNER;
        const float* Wo  = W_out + (size_t)l * D_MODEL * D_INNER;

        dim3 gxp(XP_COLS / BN, mtiles);
        gemm_nt_wmma<<<gxp, 256, 0, stream>>>(x, Wi, xp, M_TOK, XP_COLS, D_MODEL,
                                              0, nullptr, nullptr);
        dim3 gdt(D_INNER / BN, mtiles);
        gemm_nt_wmma<<<gdt, 256, 0, stream>>>(x, Wd, dtc, M_TOK, D_INNER, D_MODEL,
                                              1, bd, db);
        dim3 gbc(1, mtiles);   // N=16 handled by bounds checks
        gemm_nt_wmma<<<gbc, 256, 0, stream>>>(x, WBl, Bm, M_TOK, D_STATE, D_MODEL,
                                              0, nullptr, nullptr);
        gemm_nt_wmma<<<gbc, 256, 0, stream>>>(x, WCl, Cm, M_TOK, D_STATE, D_MODEL,
                                              0, nullptr, nullptr);

        scan_pass1<<<scanBlocks, 256, 0, stream>>>(xp, dtc, Bm, Al, Fbuf, H0b);
        scan_pass2<<<combBlocks, 256, 0, stream>>>(Fbuf, H0b, HSb);
        scan_pass3<<<scanBlocks, 256, 0, stream>>>(xp, dtc, Bm, Cm, Al, Dl, HSb, yg);

        dim3 gout(D_MODEL / BN, mtiles);
        gemm_nt_wmma<<<gout, 256, 0, stream>>>(yg, Wo, proj, M_TOK, D_MODEL, D_INNER,
                                               0, nullptr, nullptr);

        ln_kernel<<<M_TOK, 256, 0, stream>>>(proj, x, ln_g + (size_t)l * D_MODEL,
                                             ln_b + (size_t)l * D_MODEL, x);
    }

    // final LN (no residual), in-place
    ln_kernel<<<M_TOK, 256, 0, stream>>>(x, nullptr, fin_g, fin_b, x);

    // logits = x @ emb^T  : the 201-GFLOP GEMM, straight into d_out
    dim3 glog(VOCAB / BN, mtiles);
    gemm_nt_wmma<<<glog, 256, 0, stream>>>(x, emb, out, M_TOK, VOCAB, D_MODEL,
                                           0, nullptr, nullptr);
}
